// GroupedExperts_37323265802607
// MI455X (gfx1250) — matrix operand order repaired and checked by reading, compiled, not run on hardware
//
#include <hip/hip_runtime.h>
#include <hip/hip_bf16.h>

// Grouped MoE SwiGLU: out = (silu(x@w1[e]) * (x@w3[e])) @ w2[e], contiguous expert groups.
// bf16 WMMA (v_wmma_f32_16x16x32_bf16) with f32 accumulation.
//   Kernel 1: fused up+gate GEMM + SwiGLU epilogue -> h (bf16) in d_ws.
//   Kernel 2: down GEMM h @ w2 -> f32 out.
// Both kernels: ping-pong LDS double buffering, register prefetch of next K-tile,
// packed f32->bf16 conversion (v_cvt_pk_bf16_f32), one barrier per K-step.

typedef __attribute__((ext_vector_type(16))) __bf16 v16bf;
typedef __attribute__((ext_vector_type(2)))  __bf16 v2bf;
typedef __attribute__((ext_vector_type(2)))  float  f32x2;
typedef __attribute__((ext_vector_type(8)))  float  v8f;

#define T_TOK 16384
#define DIM   2048
#define HID   1024

static __device__ __forceinline__ unsigned int pk_bf16(float a, float b) {
    // two f32 -> packed 2xbf16 (RNE); lowers to v_cvt_pk_bf16_f32
    f32x2 f = {a, b};
    v2bf  r = __builtin_convertvector(f, v2bf);
    union { v2bf v; unsigned int u; } cv;
    cv.v = r;
    return cv.u;
}

static __device__ __forceinline__ unsigned short f2bf(float f) {
    unsigned int u = __float_as_uint(f);
    u += 0x7FFFu + ((u >> 16) & 1u);
    return (unsigned short)(u >> 16);
}

union FragBF {
    uint4 q[2];
    v16bf v;
};

static __device__ __forceinline__ int expert_of_row(const int* __restrict__ m_sizes,
                                                    int E, int row0) {
    int cum = 0;
    for (int e = 0; e < E; ++e) {
        cum += m_sizes[e];
        if (row0 < cum) return e;
    }
    return E - 1;
}

// ---------------------------------------------------------------------------
// Kernel 1: h = silu(x @ w1[e]) * (x @ w3[e])   [T, HID], stored bf16
// Block tile 128(M) x 64(N), K-step 32. 8 waves: 4(M) x 2(N), wave tile 32x32.
// ---------------------------------------------------------------------------
__global__ __launch_bounds__(256)
void moe_up_gate_kernel(const float* __restrict__ x,
                        const float* __restrict__ w1,
                        const float* __restrict__ w3,
                        const int* __restrict__ m_sizes, int E,
                        unsigned short* __restrict__ hbuf) {
    __shared__ unsigned short As[2][128][32];  // x tile bf16 [m][k]
    __shared__ unsigned short B1[2][64][32];   // w1 tile bf16 transposed [n][k]
    __shared__ unsigned short B2[2][64][32];   // w3 tile bf16 transposed [n][k]

    const int m0   = blockIdx.x * 128;
    const int n0   = blockIdx.y * 64;
    const int tid  = threadIdx.x;
    const int lane = tid & 31;
    const int wave = tid >> 5;
    const int wm   = wave >> 1;      // 0..3
    const int wn   = wave & 1;       // 0..1
    const int g    = lane >> 4;      // K-half select
    const int ln   = lane & 15;

    const int eid = expert_of_row(m_sizes, E, m0);
    const float* w1e = w1 + (size_t)eid * DIM * HID;
    const float* w3e = w3 + (size_t)eid * DIM * HID;

    // per-thread load mapping
    const int xr = tid >> 3;              // x rows handled (4 rows, stride 32)
    const int xc = (tid & 7) << 2;        // x col quad within K-slab
    const int wn4 = (tid & 15) << 2;      // weight col quad (0..60)
    const int wkp = (tid >> 4) << 1;      // weight K pair (0..30)

    float4 px[4];
    float4 pa0, pa1, pb0, pb1;

    auto load_tiles = [&](int kt) {
#pragma unroll
        for (int i = 0; i < 4; ++i)
            px[i] = *(const float4*)&x[(size_t)(m0 + xr + i * 32) * DIM + kt + xc];
        pa0 = *(const float4*)&w1e[(size_t)(kt + wkp)     * HID + n0 + wn4];
        pa1 = *(const float4*)&w1e[(size_t)(kt + wkp + 1) * HID + n0 + wn4];
        pb0 = *(const float4*)&w3e[(size_t)(kt + wkp)     * HID + n0 + wn4];
        pb1 = *(const float4*)&w3e[(size_t)(kt + wkp + 1) * HID + n0 + wn4];
    };

    auto store_tiles = [&](int buf) {
#pragma unroll
        for (int i = 0; i < 4; ++i) {
            uint2 s = {pk_bf16(px[i].x, px[i].y), pk_bf16(px[i].z, px[i].w)};
            *(uint2*)&As[buf][xr + i * 32][xc] = s;
        }
        const float* a0 = (const float*)&pa0;
        const float* a1 = (const float*)&pa1;
        const float* b0 = (const float*)&pb0;
        const float* b1 = (const float*)&pb1;
#pragma unroll
        for (int j = 0; j < 4; ++j) {
            *(unsigned int*)&B1[buf][wn4 + j][wkp] = pk_bf16(a0[j], a1[j]);
            *(unsigned int*)&B2[buf][wn4 + j][wkp] = pk_bf16(b0[j], b1[j]);
        }
    };

    v8f zero = {};
    v8f acc1[2][2], acc2[2][2];
#pragma unroll
    for (int i = 0; i < 2; ++i)
#pragma unroll
        for (int j = 0; j < 2; ++j) { acc1[i][j] = zero; acc2[i][j] = zero; }

    load_tiles(0);
    store_tiles(0);
    __syncthreads();

    for (int kt = 0; kt < DIM; kt += 32) {
        const int cur = (kt >> 5) & 1;
        const int nxt = cur ^ 1;
        const bool more = (kt + 32) < DIM;

        if (more) load_tiles(kt + 32);   // global loads in flight during wmma

#pragma unroll
        for (int mt = 0; mt < 2; ++mt) {
            FragBF a;
            const int row = wm * 32 + mt * 16 + ln;
            a.q[0] = *(const uint4*)&As[cur][row][g * 16];
            a.q[1] = *(const uint4*)&As[cur][row][g * 16 + 8];
#pragma unroll
            for (int nt = 0; nt < 2; ++nt) {
                const int col = wn * 32 + nt * 16 + ln;
                FragBF b;
                b.q[0] = *(const uint4*)&B1[cur][col][g * 16];
                b.q[1] = *(const uint4*)&B1[cur][col][g * 16 + 8];
                acc1[mt][nt] = __builtin_amdgcn_wmma_f32_16x16x32_bf16(
                    false, a.v, false, b.v, (short)0, acc1[mt][nt], false, false);
                FragBF c;
                c.q[0] = *(const uint4*)&B2[cur][col][g * 16];
                c.q[1] = *(const uint4*)&B2[cur][col][g * 16 + 8];
                acc2[mt][nt] = __builtin_amdgcn_wmma_f32_16x16x32_bf16(
                    false, a.v, false, c.v, (short)0, acc2[mt][nt], false, false);
            }
        }

        if (more) store_tiles(nxt);
        __syncthreads();
    }

    // --- epilogue: SwiGLU, write bf16 h (fast rcp instead of IEEE div)
#pragma unroll
    for (int mt = 0; mt < 2; ++mt) {
#pragma unroll
        for (int nt = 0; nt < 2; ++nt) {
            const int colg = n0 + wn * 32 + nt * 16 + ln;
#pragma unroll
            for (int r = 0; r < 8; ++r) {
                const int rowg = m0 + wm * 32 + mt * 16 + g * 8 + r;
                float up = acc1[mt][nt][r];
                float gt = acc2[mt][nt][r];
                float s  = up * __builtin_amdgcn_rcpf(1.0f + __expf(-up));
                hbuf[(size_t)rowg * HID + colg] = f2bf(s * gt);
            }
        }
    }
}

// ---------------------------------------------------------------------------
// Kernel 2: out = h @ w2[e]   [T, DIM] f32
// Block tile 128(M) x 128(N), K-step 32. 8 waves: 4(M) x 2(N), wave tile 32x64.
// ---------------------------------------------------------------------------
__global__ __launch_bounds__(256)
void moe_down_kernel(const unsigned short* __restrict__ hbuf,
                     const float* __restrict__ w2,
                     const int* __restrict__ m_sizes, int E,
                     float* __restrict__ out) {
    __shared__ unsigned short As[2][128][32];  // h tile bf16 [m][k]
    __shared__ unsigned short Bs[2][128][32];  // w2 tile bf16 transposed [n][k]

    const int m0   = blockIdx.x * 128;
    const int n0   = blockIdx.y * 128;
    const int tid  = threadIdx.x;
    const int lane = tid & 31;
    const int wave = tid >> 5;
    const int wm   = wave >> 1;      // 0..3 -> 32 rows
    const int wn   = wave & 1;       // 0..1 -> 64 cols
    const int g    = lane >> 4;
    const int ln   = lane & 15;

    const int eid = expert_of_row(m_sizes, E, m0);
    const float* w2e = w2 + (size_t)eid * HID * DIM;

    // per-thread load mapping
    const int hr = tid >> 1;              // h row (2 uint4 per row pair)
    const int hc = (tid & 1) << 4;        // 16-ushort half of 32-wide slab
    // weights: 32 n-quads x 16 k-pairs = 512 units, 2 per thread
    const int wn4a = (tid & 31) << 2;
    const int wkpa = (tid >> 5) << 1;     // 0..14
    const int wkpb = wkpa + 16;           // 16..30

    uint4  ph[2];
    float4 pw[4];

    auto load_tiles = [&](int kt) {
        ph[0] = *(const uint4*)&hbuf[(size_t)(m0 + hr) * HID + kt + hc];
        ph[1] = *(const uint4*)&hbuf[(size_t)(m0 + hr) * HID + kt + hc + 8];
        pw[0] = *(const float4*)&w2e[(size_t)(kt + wkpa)     * DIM + n0 + wn4a];
        pw[1] = *(const float4*)&w2e[(size_t)(kt + wkpa + 1) * DIM + n0 + wn4a];
        pw[2] = *(const float4*)&w2e[(size_t)(kt + wkpb)     * DIM + n0 + wn4a];
        pw[3] = *(const float4*)&w2e[(size_t)(kt + wkpb + 1) * DIM + n0 + wn4a];
    };

    auto store_tiles = [&](int buf) {
        *(uint4*)&As[buf][hr][hc]     = ph[0];
        *(uint4*)&As[buf][hr][hc + 8] = ph[1];
        const float* w0 = (const float*)&pw[0];
        const float* w1p = (const float*)&pw[1];
        const float* w2p = (const float*)&pw[2];
        const float* w3p = (const float*)&pw[3];
#pragma unroll
        for (int j = 0; j < 4; ++j) {
            *(unsigned int*)&Bs[buf][wn4a + j][wkpa] = pk_bf16(w0[j], w1p[j]);
            *(unsigned int*)&Bs[buf][wn4a + j][wkpb] = pk_bf16(w2p[j], w3p[j]);
        }
    };

    v8f zero = {};
    v8f acc[2][4];
#pragma unroll
    for (int i = 0; i < 2; ++i)
#pragma unroll
        for (int j = 0; j < 4; ++j) acc[i][j] = zero;

    load_tiles(0);
    store_tiles(0);
    __syncthreads();

    for (int kt = 0; kt < HID; kt += 32) {
        const int cur = (kt >> 5) & 1;
        const int nxt = cur ^ 1;
        const bool more = (kt + 32) < HID;

        if (more) load_tiles(kt + 32);

#pragma unroll
        for (int mt = 0; mt < 2; ++mt) {
            FragBF a;
            const int row = wm * 32 + mt * 16 + ln;
            a.q[0] = *(const uint4*)&As[cur][row][g * 16];
            a.q[1] = *(const uint4*)&As[cur][row][g * 16 + 8];
#pragma unroll
            for (int nt = 0; nt < 4; ++nt) {
                const int col = wn * 64 + nt * 16 + ln;
                FragBF b;
                b.q[0] = *(const uint4*)&Bs[cur][col][g * 16];
                b.q[1] = *(const uint4*)&Bs[cur][col][g * 16 + 8];
                acc[mt][nt] = __builtin_amdgcn_wmma_f32_16x16x32_bf16(
                    false, a.v, false, b.v, (short)0, acc[mt][nt], false, false);
            }
        }

        if (more) store_tiles(nxt);
        __syncthreads();
    }

    // --- epilogue: f32 out
#pragma unroll
    for (int mt = 0; mt < 2; ++mt) {
#pragma unroll
        for (int nt = 0; nt < 4; ++nt) {
            const int colg = n0 + wn * 64 + nt * 16 + ln;
#pragma unroll
            for (int r = 0; r < 8; ++r) {
                const int rowg = m0 + wm * 32 + mt * 16 + g * 8 + r;
                out[(size_t)rowg * DIM + colg] = acc[mt][nt][r];
            }
        }
    }
}

extern "C" void kernel_launch(void* const* d_in, const int* in_sizes, int n_in,
                              void* d_out, int out_size, void* d_ws, size_t ws_size,
                              hipStream_t stream) {
    const float* x       = (const float*)d_in[0];
    const float* w1      = (const float*)d_in[1];
    const float* w2      = (const float*)d_in[2];
    const float* w3      = (const float*)d_in[3];
    const int*   m_sizes = (const int*)d_in[4];
    const int    E       = in_sizes[4];

    unsigned short* hbuf = (unsigned short*)d_ws;   // T_TOK * HID bf16 (32 MB)
    float* out = (float*)d_out;

    dim3 block(256);
    dim3 grid1(T_TOK / 128, HID / 64);    // 128 x 16
    moe_up_gate_kernel<<<grid1, block, 0, stream>>>(x, w1, w3, m_sizes, E, hbuf);

    dim3 grid2(T_TOK / 128, DIM / 128);   // 128 x 16
    moe_down_kernel<<<grid2, block, 0, stream>>>(hbuf, w2, m_sizes, E, out);
}
